// GIN_26877905339098
// MI455X (gfx1250) — matrix operand order repaired and checked by reading, not compiled, not run on hardware
//
#include <hip/hip_runtime.h>

#define HD 128
#define WT_STRIDE 136   // padded bf16 row stride in LDS (272B rows -> conflict-free b128 reads)
#define USE_ASM_PK_BF16 1

typedef __attribute__((ext_vector_type(16))) __bf16          v16bf;
typedef __attribute__((ext_vector_type(8)))  float           v8f;
typedef __attribute__((ext_vector_type(16))) unsigned short  v16u;
typedef __attribute__((ext_vector_type(8)))  unsigned short  v8u;
typedef __attribute__((ext_vector_type(8)))  unsigned int    v8ui;

__device__ __forceinline__ unsigned short f2bf(float x) {
    unsigned u = __float_as_uint(x);
    unsigned r = 0x7FFFu + ((u >> 16) & 1u);
    return (unsigned short)((u + r) >> 16);
}

// pack two f32 -> two bf16 in one dword via the CDNA5 packed converter
__device__ __forceinline__ unsigned pk_bf16(float lo, float hi) {
#if USE_ASM_PK_BF16
    unsigned r;
    asm volatile("v_cvt_pk_bf16_f32 %0, %1, %2" : "=v"(r) : "v"(lo), "v"(hi));
    return r;
#else
    return (unsigned)f2bf(lo) | ((unsigned)f2bf(hi) << 16);
#endif
}

// ---------------- copy: z = h (one float4 per thread) ----------------
__global__ void GIN_copy_f4(const float* __restrict__ in, float* __restrict__ out, long n4) {
    long i = (long)blockIdx.x * blockDim.x + threadIdx.x;
    if (i < n4) ((float4*)out)[i] = ((const float4*)in)[i];
}

// ------- edge aggregation: z[dst[e]] += h[src[e]]  (wave per edge, L2-resident) -------
__global__ void GIN_edge_agg(const float* __restrict__ h, const int* __restrict__ src,
                             const int* __restrict__ dst, float* __restrict__ z, int E) {
    long gid = (long)blockIdx.x * blockDim.x + threadIdx.x;
    int e = (int)(gid >> 5);
    if (e >= E) return;
    int c = ((int)gid & 31) * 4;
    int s = src[e], d = dst[e];          // wave-uniform
    float4 v = *(const float4*)(h + (long)s * HD + c);
    float* p = z + (long)d * HD + c;
    atomicAdd(p + 0, v.x);
    atomicAdd(p + 1, v.y);
    atomicAdd(p + 2, v.z);
    atomicAdd(p + 3, v.w);
}

// ------- edge feature: ef[e] = h[ia[e]] * h[ib[e]] (elementwise) -------
__global__ void GIN_edge_prod(const float* __restrict__ h, const int* __restrict__ ia,
                              const int* __restrict__ ib, float* __restrict__ ef, int E) {
    long gid = (long)blockIdx.x * blockDim.x + threadIdx.x;
    int e = (int)(gid >> 5);
    if (e >= E) return;
    int c = ((int)gid & 31) * 4;
    int a = ia[e], b = ib[e];
    float4 va = *(const float4*)(h + (long)a * HD + c);
    float4 vb = *(const float4*)(h + (long)b * HD + c);
    float4 o;
    o.x = va.x * vb.x; o.y = va.y * vb.y; o.z = va.z * vb.z; o.w = va.w * vb.w;
    *(float4*)(ef + (long)e * HD + c) = o;
}

// ------- C[M,128] = act(A[M,128] @ W[128,128] + bias), bf16 WMMA, f32 accum -------
__global__ __launch_bounds__(256)
void GIN_gemm_wmma(const float* __restrict__ A, const float* __restrict__ W,
                   const float* __restrict__ bias, float* __restrict__ C,
                   int M, int relu) {
    __shared__ __align__(16) unsigned short Wt[HD * WT_STRIDE]; // W^T in bf16, padded rows

    int tid = threadIdx.x;
    // stage W^T (n-major) into LDS as bf16; W is [k][n] row-major.
    // Each thread handles (n, k-pair): one packed cvt + one dword store.
    for (int i = tid; i < HD * (HD / 2); i += 256) {
        int n  = i >> 6;          // 0..127
        int k2 = i & 63;          // k-pair index 0..63
        float w0 = W[(2 * k2) * HD + n];
        float w1 = W[(2 * k2 + 1) * HD + n];
        *(unsigned*)&Wt[n * WT_STRIDE + 2 * k2] = pk_bf16(w0, w1);
    }
    __syncthreads();

    int wave = tid >> 5;
    int lane = tid & 31;
    int rowTile = blockIdx.x * 8 + wave;     // 16 rows per wave
    if (rowTile * 16 >= M) return;           // wave-uniform exit (EXEC stays all-ones)
    int row0 = rowTile * 16;
    int m    = lane & 15;
    int grp  = lane >> 4;
    const float* Arow = A + (long)(row0 + m) * HD;

    v8f acc[8];
    #pragma unroll
    for (int nt = 0; nt < 8; ++nt)
        #pragma unroll
        for (int j = 0; j < 8; ++j) acc[nt][j] = 0.0f;

    #pragma unroll
    for (int ks = 0; ks < 4; ++ks) {
        // A fragment (ISA layout): halves 0-7  <- K = ks*32 + grp*8 + 0..7
        //                          halves 8-15 <- K = ks*32 + 16 + grp*8 + 0..7
        int kb = ks * 32 + grp * 8;
        float4 a0 = *(const float4*)(Arow + kb);
        float4 a1 = *(const float4*)(Arow + kb + 4);
        float4 a2 = *(const float4*)(Arow + kb + 16);
        float4 a3 = *(const float4*)(Arow + kb + 20);
        v8ui au;
        au[0] = pk_bf16(a0.x, a0.y);
        au[1] = pk_bf16(a0.z, a0.w);
        au[2] = pk_bf16(a1.x, a1.y);
        au[3] = pk_bf16(a1.z, a1.w);
        au[4] = pk_bf16(a2.x, a2.y);
        au[5] = pk_bf16(a2.z, a2.w);
        au[6] = pk_bf16(a3.x, a3.y);
        au[7] = pk_bf16(a3.z, a3.w);
        v16bf afrag = __builtin_bit_cast(v16bf, au);

        int kw = ks * 32 + grp * 8;    // B fragment: lane n reads K = kw+0..7 and kw+16..23 from W^T
        #pragma unroll
        for (int nt = 0; nt < 8; ++nt) {
            int n = nt * 16 + (lane & 15);
            const unsigned short* bp = &Wt[n * WT_STRIDE + kw];
            v8u b0 = *(const v8u*)bp;
            v8u b1 = *(const v8u*)(bp + 16);
            v16u bu = __builtin_shufflevector(b0, b1,
                         0,1,2,3,4,5,6,7,8,9,10,11,12,13,14,15);
            v16bf bfrag = __builtin_bit_cast(v16bf, bu);
            acc[nt] = __builtin_amdgcn_wmma_f32_16x16x32_bf16(
                false, afrag, false, bfrag, (short)0, acc[nt], false, false);
        }
    }

    // epilogue: C/D layout -> VGPR r holds row (r + grp*8), lanes%16 = col within tile
    #pragma unroll
    for (int nt = 0; nt < 8; ++nt) {
        int col = nt * 16 + (lane & 15);
        float b = bias ? bias[col] : 0.0f;
        #pragma unroll
        for (int r = 0; r < 8; ++r) {
            float v = acc[nt][r] + b;
            if (relu) v = fmaxf(v, 0.0f);
            int row = row0 + r + grp * 8;
            C[(long)row * HD + col] = v;
        }
    }
}

// ------- out[row] = dot(E[row,:], w) + b[0]  (wave per row, shfl reduction) -------
__global__ void GIN_dot_p3(const float* __restrict__ Ein, const float* __restrict__ w,
                           const float* __restrict__ b, float* __restrict__ out, int M) {
    long gid = (long)blockIdx.x * blockDim.x + threadIdx.x;
    int row = (int)(gid >> 5);
    if (row >= M) return;
    int lane = (int)gid & 31;
    float4 v  = *(const float4*)(Ein + (long)row * HD + lane * 4);
    float4 ww = *(const float4*)(w + lane * 4);
    float s = v.x * ww.x + v.y * ww.y + v.z * ww.z + v.w * ww.w;
    s += __shfl_xor(s, 16);
    s += __shfl_xor(s, 8);
    s += __shfl_xor(s, 4);
    s += __shfl_xor(s, 2);
    s += __shfl_xor(s, 1);
    if (lane == 0) out[row] = s + b[0];
}

extern "C" void kernel_launch(void* const* d_in, const int* in_sizes, int n_in,
                              void* d_out, int out_size, void* d_ws, size_t ws_size,
                              hipStream_t stream) {
    const float* x       = (const float*)d_in[0];
    const int*   src     = (const int*)d_in[1];
    const int*   dst     = (const int*)d_in[2];
    const int*   pos_src = (const int*)d_in[3];
    const int*   pos_dst = (const int*)d_in[4];
    const int*   neg_src = (const int*)d_in[5];
    const int*   neg_dst = (const int*)d_in[6];
    const float* l0w1 = (const float*)d_in[7];
    const float* l0w2 = (const float*)d_in[8];
    const float* l1w1 = (const float*)d_in[9];
    const float* l1w2 = (const float*)d_in[10];
    const float* l2w1 = (const float*)d_in[11];
    const float* l2w2 = (const float*)d_in[12];
    const float* p1w  = (const float*)d_in[13];
    const float* p1b  = (const float*)d_in[14];
    const float* p2w  = (const float*)d_in[15];
    const float* p2b  = (const float*)d_in[16];
    const float* p3w  = (const float*)d_in[17];
    const float* p3b  = (const float*)d_in[18];
    float* out = (float*)d_out;

    const int N  = in_sizes[0] / HD;   // 100000
    const int E  = in_sizes[1];        // 1600000
    const int EP = in_sizes[3];        // 200000

    float* hA = (float*)d_ws;                 // [N,128]
    float* zB = hA + (size_t)N * HD;          // [N,128]
    float* tC = zB + (size_t)N * HD;          // [N,128]
    float* ef = tC + (size_t)N * HD;          // [EP,128]
    float* et = ef + (size_t)EP * HD;         // [EP,128]

    long n4 = (long)N * HD / 4;
    int copyGrid = (int)((n4 + 255) / 256);
    int aggGrid  = (int)(((long)E * 32 + 255) / 256);
    long e32 = (long)EP * 32;
    int edgeGrid = (int)((e32 + 255) / 256);

    auto gemm = [&](const float* A, const float* W, const float* bias, float* Cc,
                    int M, int relu) {
        int grid = ((M + 15) / 16 + 7) / 8;
        GIN_gemm_wmma<<<grid, 256, 0, stream>>>(A, W, bias, Cc, M, relu);
    };

    // ---- Layer 0: z = x + segsum(x[src]); t = relu(z@w1); h = relu(t@w2)
    GIN_copy_f4<<<copyGrid, 256, 0, stream>>>(x, zB, n4);
    GIN_edge_agg<<<aggGrid, 256, 0, stream>>>(x, src, dst, zB, E);
    gemm(zB, l0w1, nullptr, tC, N, 1);
    gemm(tC, l0w2, nullptr, hA, N, 1);

    // ---- Layer 1
    GIN_copy_f4<<<copyGrid, 256, 0, stream>>>(hA, zB, n4);
    GIN_edge_agg<<<aggGrid, 256, 0, stream>>>(hA, src, dst, zB, E);
    gemm(zB, l1w1, nullptr, tC, N, 1);
    gemm(tC, l1w2, nullptr, hA, N, 1);

    // ---- Layer 2 (no final relu) -> final h in zB
    GIN_copy_f4<<<copyGrid, 256, 0, stream>>>(hA, zB, n4);
    GIN_edge_agg<<<aggGrid, 256, 0, stream>>>(hA, src, dst, zB, E);
    gemm(zB, l2w1, nullptr, tC, N, 1);
    gemm(tC, l2w2, nullptr, zB, N, 0);
    const float* hfin = zB;

    // ---- positive edges -> out[0..EP)
    GIN_edge_prod<<<edgeGrid, 256, 0, stream>>>(hfin, pos_src, pos_dst, ef, EP);
    gemm(ef, p1w, p1b, et, EP, 1);
    gemm(et, p2w, p2b, ef, EP, 1);
    GIN_dot_p3<<<edgeGrid, 256, 0, stream>>>(ef, p3w, p3b, out, EP);

    // ---- negative edges -> out[EP..2EP)
    GIN_edge_prod<<<edgeGrid, 256, 0, stream>>>(hfin, neg_src, neg_dst, ef, EP);
    gemm(ef, p1w, p1b, et, EP, 1);
    gemm(et, p2w, p2b, ef, EP, 1);
    GIN_dot_p3<<<edgeGrid, 256, 0, stream>>>(ef, p3w, p3b, out + EP, EP);
}
